// NotebookModel_61778809585845
// MI455X (gfx1250) — compile-verified, matrix operand order checked by reading, not run on hardware
//
#include <hip/hip_runtime.h>
#include <hip/hip_bf16.h>

// ---------------------------------------------------------------------------
// Model dims
// ---------------------------------------------------------------------------
#define DMODEL 768
#define NHEAD  12
#define DHEAD  64
#define FFDIM  3072
#define BS     4
#define NC     127
#define CLEN   64
#define S2     128          // NC+1
#define M1     (BS*NC*CLEN) // 32512 token rows
#define M2     (BS*S2)      // 512 stage-3 rows
#define FFCHUNK 4064        // 32512 / 8

// GEMM block tile
#define BM 128
#define BN 64
#define BK 32
#define LDST 40             // padded LDS row stride (halves), 80B, 16B-aligned

typedef __attribute__((ext_vector_type(16))) _Float16 v16h;
typedef __attribute__((ext_vector_type(8)))  _Float16 v8h;
typedef __attribute__((ext_vector_type(8)))  float    v8f;
typedef __attribute__((ext_vector_type(4)))  int      v4i;

// ---------------------------------------------------------------------------
// CDNA5 async global->LDS path (ASYNCcnt) if the toolchain declares it
// ---------------------------------------------------------------------------
#if defined(__has_builtin)
#if __has_builtin(__builtin_amdgcn_global_load_async_to_lds_b128)
#define NB_ASYNC_LDS 1
#endif
#endif

#ifdef NB_ASYNC_LDS
typedef __attribute__((address_space(1))) v4i* nb_gp;
typedef __attribute__((address_space(3))) v4i* nb_lp;
#endif

__device__ __forceinline__ void cp16(const _Float16* g, _Float16* l) {
#ifdef NB_ASYNC_LDS
    v4i* gg = (v4i*)g;            // reinterpret pointee (generic AS)
    v4i* ll = (v4i*)l;
    __builtin_amdgcn_global_load_async_to_lds_b128((nb_gp)gg, (nb_lp)ll, 0, 0);
#else
    *(v8h*)l = *(const v8h*)g;
#endif
}

__device__ __forceinline__ void cp_wait() {
#ifdef NB_ASYNC_LDS
#if __has_builtin(__builtin_amdgcn_s_wait_asynccnt)
    __builtin_amdgcn_s_wait_asynccnt(0);
#else
    asm volatile("s_wait_asynccnt 0x0" ::: "memory");
#endif
#endif
}

// ---------------------------------------------------------------------------
// Device helpers
// ---------------------------------------------------------------------------
__device__ __forceinline__ float block_reduce_256(float v, float* red) {
    int tid = threadIdx.x;
    red[tid] = v;
    __syncthreads();
    for (int off = 128; off > 0; off >>= 1) {
        if (tid < off) red[tid] += red[tid + off];
        __syncthreads();
    }
    float r = red[0];
    __syncthreads();
    return r;
}

__device__ __forceinline__ float gelu_exact(float x) {
    return 0.5f * x * (1.0f + erff(x * 0.70710678118654752f));
}

__device__ __forceinline__ float lrelu(float x) {
    return x > 0.0f ? x : 0.01f * x;
}

// Build a 16x32 f16 WMMA A/B fragment from a row-major tile pointer p that
// already includes (row * stride + half*8):  f[0..7]=cols +0..7, f[8..15]=+16..23
__device__ __forceinline__ v16h frag_ld(const _Float16* p) {
    v8h lo = *(const v8h*)p;
    v8h hi = *(const v8h*)(p + 16);
    v16h f;
#pragma unroll
    for (int i = 0; i < 8; ++i) { f[i] = lo[i]; f[i + 8] = hi[i]; }
    return f;
}

// Fragment directly from global memory (row-major, leading dim ld, row clamp)
__device__ __forceinline__ v16h load_frag_g(const _Float16* __restrict__ A,
                                            int row0, int Mtot, int ld, int k0) {
    int lane = threadIdx.x & 31;
    int r = row0 + (lane & 15);
    if (r >= Mtot) r = Mtot - 1;
    int half = lane >> 4;
    return frag_ld(A + (size_t)r * ld + k0 + half * 8);
}

// ---------------------------------------------------------------------------
// Weight transpose+downconvert: W (K x N, f32) -> Wt (N x K, f16)
// ---------------------------------------------------------------------------
__global__ void nb_transpose_f16(const float* __restrict__ W, _Float16* __restrict__ Wt,
                                 int K, int N) {
    size_t e = (size_t)blockIdx.x * blockDim.x + threadIdx.x;
    if (e < (size_t)K * N) {
        int k = (int)(e / N), n = (int)(e % N);
        Wt[(size_t)n * K + k] = (_Float16)W[e];
    }
}

// ---------------------------------------------------------------------------
// LDS-tiled WMMA GEMM: C[M,N] = act(A[M,K](f16) @ Bt[N,K](f16)^T + bias)
// Block (256 thr, 8 waves) computes a 128x64 C tile; A/B K-slabs double-
// buffered in LDS (async global->LDS when available); wave w owns rows
// [w*16, w*16+16).  act: 0=none, 1=gelu(exact), 2=tanh
// ---------------------------------------------------------------------------
__global__ void nb_gemm_wmma(const _Float16* __restrict__ A,
                             const _Float16* __restrict__ Bt,
                             const float* __restrict__ bias,
                             float* __restrict__ Cf,
                             _Float16* __restrict__ Ch,
                             int M, int N, int K, int act) {
    __shared__ __align__(16) _Float16 As[2][BM * LDST];
    __shared__ __align__(16) _Float16 Bs[2][BN * LDST];

    int ntiles = N / BN;
    int m0 = (blockIdx.x / ntiles) * BM;
    int n0 = (blockIdx.x % ntiles) * BN;
    int tid = threadIdx.x, wv = tid >> 5, lane = tid & 31;
    int rm = lane & 15, half = lane >> 4;

    // stage K-slab k0 into buffer p (3 x 16B per thread)
    auto stage = [&](int k0, int p) {
#pragma unroll
        for (int it = 0; it < 2; ++it) {
            int chunk = tid + (it << 8);          // 0..511 : A is 128 rows x 4 chunks
            int row = chunk >> 2, c = chunk & 3;
            int gr = m0 + row; if (gr >= M) gr = M - 1;
            cp16(A + (size_t)gr * K + k0 + c * 8, &As[p][row * LDST + c * 8]);
        }
        {
            int row = tid >> 2, c = tid & 3;      // B is 64 rows x 4 chunks
            cp16(Bt + (size_t)(n0 + row) * K + k0 + c * 8, &Bs[p][row * LDST + c * 8]);
        }
    };

    v8f acc[4] = {v8f{}, v8f{}, v8f{}, v8f{}};
    stage(0, 0);
    cp_wait();
    __syncthreads();

    int p = 0;
    for (int k0 = 0; k0 < K; k0 += BK) {
        if (k0 + BK < K) stage(k0 + BK, p ^ 1);
        v16h af = frag_ld(&As[p][(wv * 16 + rm) * LDST + half * 8]);
#pragma unroll
        for (int t = 0; t < 4; ++t) {
            v16h bf = frag_ld(&Bs[p][(t * 16 + rm) * LDST + half * 8]);
            acc[t] = __builtin_amdgcn_wmma_f32_16x16x32_f16(
                false, af, false, bf, (short)0, acc[t], false, false);
        }
        cp_wait();
        __syncthreads();
        p ^= 1;
    }

#pragma unroll
    for (int t = 0; t < 4; ++t) {
        int n = n0 + t * 16 + rm;
        float bv = bias ? bias[n] : 0.0f;
#pragma unroll
        for (int r = 0; r < 8; ++r) {
            int m = m0 + wv * 16 + r + half * 8;
            if (m < M) {
                float v = acc[t][r] + bv;
                if (act == 1)      v = gelu_exact(v);
                else if (act == 2) v = tanhf(v);
                if (Cf) Cf[(size_t)m * N + n] = v;
                if (Ch) Ch[(size_t)m * N + n] = (_Float16)v;
            }
        }
    }
}

// ---------------------------------------------------------------------------
// Embedding + LayerNorm: row -> LN(word_emb[ids[row]] + pos_emb[pos]) ; 256 thr
// pos = posmod ? row % posmod : 0 ; output row = row * out_mult
// ---------------------------------------------------------------------------
__global__ void nb_embed_ln(const int* __restrict__ ids,
                            const float* __restrict__ word_emb,
                            const float* __restrict__ pos_emb,
                            const float* __restrict__ g, const float* __restrict__ b,
                            float* __restrict__ out32, _Float16* __restrict__ out16,
                            int posmod, int out_mult) {
    __shared__ float red[256];
    int row = blockIdx.x, tid = threadIdx.x;
    int id  = ids[row];
    int pos = posmod ? (row % posmod) : 0;
    float v[3], s = 0.0f;
#pragma unroll
    for (int i = 0; i < 3; ++i) {
        int d = tid + (i << 8);
        float x = word_emb[(size_t)id * DMODEL + d] + pos_emb[(size_t)pos * DMODEL + d];
        v[i] = x; s += x;
    }
    float mean = block_reduce_256(s, red) * (1.0f / DMODEL);
    s = 0.0f;
#pragma unroll
    for (int i = 0; i < 3; ++i) { v[i] -= mean; s += v[i] * v[i]; }
    float var  = block_reduce_256(s, red) * (1.0f / DMODEL);
    float rstd = rsqrtf(var + 1e-12f);
    size_t orow = (size_t)row * out_mult;
#pragma unroll
    for (int i = 0; i < 3; ++i) {
        int d = tid + (i << 8);
        float y = v[i] * rstd * g[d] + b[d];
        if (out32) out32[orow * DMODEL + d] = y;
        if (out16) out16[orow * DMODEL + d] = (_Float16)y;
    }
}

// ---------------------------------------------------------------------------
// (X + Y) -> LayerNorm -> out32 / out16. Y row = (ymod ? row % ymod : row).
// ---------------------------------------------------------------------------
__global__ void nb_add_ln(const float* __restrict__ X, const float* __restrict__ Y,
                          const float* __restrict__ g, const float* __restrict__ b,
                          float* __restrict__ out32, _Float16* __restrict__ out16,
                          int ymod) {
    __shared__ float red[256];
    int row = blockIdx.x, tid = threadIdx.x;
    int yrow = ymod ? (row % ymod) : row;
    float v[3], s = 0.0f;
#pragma unroll
    for (int i = 0; i < 3; ++i) {
        int d = tid + (i << 8);
        float x = X[(size_t)row * DMODEL + d];
        if (Y) x += Y[(size_t)yrow * DMODEL + d];
        v[i] = x; s += x;
    }
    float mean = block_reduce_256(s, red) * (1.0f / DMODEL);
    s = 0.0f;
#pragma unroll
    for (int i = 0; i < 3; ++i) { v[i] -= mean; s += v[i] * v[i]; }
    float var  = block_reduce_256(s, red) * (1.0f / DMODEL);
    float rstd = rsqrtf(var + 1e-12f);
#pragma unroll
    for (int i = 0; i < 3; ++i) {
        int d = tid + (i << 8);
        float y = v[i] * rstd * g[d] + b[d];
        if (out32) out32[(size_t)row * DMODEL + d] = y;
        if (out16) out16[(size_t)row * DMODEL + d] = (_Float16)y;
    }
}

// ---------------------------------------------------------------------------
// Softmax attention, one block = (seq, head), blockDim = S (64 or 128).
// Q/K/V f16 rows of width 768 (col = head*64 + d). mask stride per seq = S.
// ---------------------------------------------------------------------------
__global__ void nb_attention(const _Float16* __restrict__ Qh,
                             const _Float16* __restrict__ Kh,
                             const _Float16* __restrict__ Vh,
                             const int* __restrict__ mask,
                             _Float16* __restrict__ Oh, int S) {
    extern __shared__ float smem[];      // S*64 K + S*64 V
    float* Ks = smem;
    float* Vs = smem + S * DHEAD;
    int seq = blockIdx.x / NHEAD, hh = blockIdx.x % NHEAD;
    size_t base = (size_t)seq * S * DMODEL + hh * DHEAD;

    for (int e = threadIdx.x; e < S * DHEAD; e += blockDim.x) {
        int t = e >> 6, d = e & 63;
        Ks[e] = (float)Kh[base + (size_t)t * DMODEL + d];
        Vs[e] = (float)Vh[base + (size_t)t * DMODEL + d];
    }
    __syncthreads();

    int q = threadIdx.x;
    float qv[DHEAD];
#pragma unroll
    for (int d = 0; d < DHEAD; ++d) qv[d] = (float)Qh[base + (size_t)q * DMODEL + d];

    float sc[128];
    float mx = -1e30f;
    for (int t = 0; t < S; ++t) {
        float s = 0.0f;
#pragma unroll
        for (int d = 0; d < DHEAD; ++d) s += qv[d] * Ks[t * DHEAD + d];
        s *= 0.125f;                            // 1/sqrt(64)
        if (mask[seq * S + t] == 0) s -= 1e9f;
        sc[t] = s; mx = fmaxf(mx, s);
    }
    float den = 0.0f;
    for (int t = 0; t < S; ++t) { sc[t] = expf(sc[t] - mx); den += sc[t]; }
    float inv = 1.0f / den;
    for (int d = 0; d < DHEAD; ++d) {
        float o = 0.0f;
        for (int t = 0; t < S; ++t) o += sc[t] * Vs[t * DHEAD + d];
        Oh[base + (size_t)q * DMODEL + d] = (_Float16)(o * inv);
    }
}

// ---------------------------------------------------------------------------
// Cell aggregation: s_t = TW[t]·agg_v (TW = tanh(tok@agg_W), f32), masked
// softmax over 64 tokens, cell[d] = sum_t w_t * Tok[t][d].  block = 64 thr.
// ---------------------------------------------------------------------------
__global__ void nb_agg(const float* __restrict__ TW, const float* __restrict__ Tok,
                       const float* __restrict__ agg_v, const int* __restrict__ cell_masks,
                       float* __restrict__ X3pre) {
    __shared__ float w[CLEN];
    __shared__ float red[CLEN];
    int cell = blockIdx.x;                 // 0..507
    int b = cell / NC, c = cell % NC;
    int t = threadIdx.x;
    size_t row = (size_t)cell * CLEN + t;
    float s = 0.0f;
    for (int d = 0; d < DMODEL; ++d) s += TW[row * DMODEL + d] * agg_v[d];
    if (cell_masks[cell * CLEN + t] == 0) s = -65000.0f;
    w[t] = s;
    __syncthreads();
    float mx = w[0];
    for (int i = 1; i < CLEN; ++i) mx = fmaxf(mx, w[i]);
    float e = expf(w[t] - mx);
    red[t] = e;
    __syncthreads();
    float den = 0.0f;
    for (int i = 0; i < CLEN; ++i) den += red[i];
    __syncthreads();
    w[t] = e / den;
    __syncthreads();
    float* outrow = X3pre + ((size_t)b * S2 + 1 + c) * DMODEL;
    for (int d = t; d < DMODEL; d += CLEN) {
        float acc = 0.0f;
        for (int k = 0; k < CLEN; ++k)
            acc += w[k] * Tok[((size_t)cell * CLEN + k) * DMODEL + d];
        outrow[d] = acc;
    }
}

// ---------------------------------------------------------------------------
// cb[b][n] = fts[b] * W0[2D][n] + b0[n]
// ---------------------------------------------------------------------------
__global__ void nb_cb(const float* __restrict__ fts, const float* __restrict__ W0,
                      const float* __restrict__ b0, float* __restrict__ cb) {
    int e = blockIdx.x * blockDim.x + threadIdx.x;
    if (e < BS * 512) {
        int b = e >> 9, n = e & 511;
        cb[e] = fts[b] * W0[2 * DMODEL * 512 + n] + b0[n];
    }
}

// ---------------------------------------------------------------------------
// Pairwise head: block = (jtile, i, b). Build 16x512 f16 H tile in LDS,
// 8 waves WMMA H @ W1t (K=512, N=128), leaky_relu, fold Wt via LDS atomics.
// ---------------------------------------------------------------------------
__global__ void nb_pairwise(const float* __restrict__ A4, const float* __restrict__ B4,
                            const float* __restrict__ cb,
                            const _Float16* __restrict__ W1t,  // 128 x 512
                            const float* __restrict__ b1,
                            const float* __restrict__ Wt, const float* __restrict__ bt,
                            const int* __restrict__ label_mask,
                            float* __restrict__ out) {
    __shared__ __align__(16) _Float16 Hh[16 * 512];
    __shared__ float outacc[16];
    int jt = blockIdx.x, i = blockIdx.y, b = blockIdx.z;
    int j0 = jt * 16;
    int tid = threadIdx.x;
    if (tid < 16) outacc[tid] = 0.0f;

    const float* arow = A4 + ((size_t)b * S2 + i) * 512;
    const float* crow = cb + b * 512;
    for (int e = tid; e < 16 * 512; e += 256) {
        int m = e >> 9, n = e & 511;
        int j = j0 + m;
        float v = 0.0f;
        if (j < NC) {
            v = arow[n] + B4[((size_t)b * S2 + j + 1) * 512 + n] + crow[n];
            v = lrelu(v);
        }
        Hh[e] = (_Float16)v;
    }
    __syncthreads();

    int wv = tid >> 5, lane = tid & 31;
    int n0 = wv * 16;
    int rm = lane & 15, half = lane >> 4;
    v8f acc = {};
    for (int k0 = 0; k0 < 512; k0 += 32) {
        v16h af = frag_ld(&Hh[rm * 512 + k0 + half * 8]);
        v16h bf = load_frag_g(W1t, n0, 128, 512, k0);
        acc = __builtin_amdgcn_wmma_f32_16x16x32_f16(
            false, af, false, bf, (short)0, acc, false, false);
    }
    int n = n0 + rm;
    float b1n = b1[n], wtn = Wt[n];
#pragma unroll
    for (int r = 0; r < 8; ++r) {
        int m = r + half * 8;
        float v = lrelu(acc[r] + b1n);
        atomicAdd(&outacc[m], v * wtn);
    }
    __syncthreads();
    if (tid < 16) {
        int j = j0 + tid;
        if (j < NC) {
            size_t oidx = ((size_t)b * NC + i) * NC + j;
            float val = outacc[tid] + bt[0];
            out[oidx] = label_mask[oidx] ? -65000.0f : val;
        }
    }
}

// ---------------------------------------------------------------------------
// Host orchestration
// ---------------------------------------------------------------------------
extern "C" void kernel_launch(void* const* d_in, const int* in_sizes, int n_in,
                              void* d_out, int out_size, void* d_ws, size_t ws_size,
                              hipStream_t stream) {
    (void)in_sizes; (void)n_in; (void)out_size; (void)ws_size;
    // setup_inputs() dict order, params flattened in insertion order
    const int*   ids        = (const int*)  d_in[0];
    const int*   cell_masks = (const int*)  d_in[1];
    const int*   nb_masks   = (const int*)  d_in[2];
    const int*   label_mask = (const int*)  d_in[3];
    /* d_in[4] = pos (derived as row % 128) */
    const int*   start      = (const int*)  d_in[5];
    const float* fts        = (const float*)d_in[6];
    const float* word_emb = (const float*)d_in[7];
    const float* pos_emb  = (const float*)d_in[8];
    const float* ln_emb_g = (const float*)d_in[9];
    const float* ln_emb_b = (const float*)d_in[10];
    const float* Wq = (const float*)d_in[11]; const float* bq = (const float*)d_in[12];
    const float* Wk = (const float*)d_in[13]; const float* bk = (const float*)d_in[14];
    const float* Wv = (const float*)d_in[15]; const float* bv = (const float*)d_in[16];
    const float* Wo = (const float*)d_in[17]; const float* bo = (const float*)d_in[18];
    const float* ln1_g = (const float*)d_in[19]; const float* ln1_b = (const float*)d_in[20];
    const float* Wf1 = (const float*)d_in[21]; const float* bf1 = (const float*)d_in[22];
    const float* Wf2 = (const float*)d_in[23]; const float* bf2 = (const float*)d_in[24];
    const float* ln2_g = (const float*)d_in[25]; const float* ln2_b = (const float*)d_in[26];
    const float* agg_W = (const float*)d_in[27]; const float* agg_v = (const float*)d_in[28];
    const float* W0 = (const float*)d_in[29]; const float* b0 = (const float*)d_in[30];
    const float* W1 = (const float*)d_in[31]; const float* b1 = (const float*)d_in[32];
    const float* Wt = (const float*)d_in[33]; const float* bt = (const float*)d_in[34];

    // Workspace carve-out
    char* wsp = (char*)d_ws;
    auto carve = [&](size_t bytes) -> void* {
        void* p = (void*)wsp;
        wsp += (bytes + 255) & ~(size_t)255;
        return p;
    };
    _Float16* Wqt  = (_Float16*)carve((size_t)768 * 768 * 2);
    _Float16* Wkt  = (_Float16*)carve((size_t)768 * 768 * 2);
    _Float16* Wvt  = (_Float16*)carve((size_t)768 * 768 * 2);
    _Float16* Wot  = (_Float16*)carve((size_t)768 * 768 * 2);
    _Float16* Wf1t = (_Float16*)carve((size_t)FFDIM * 768 * 2);
    _Float16* Wf2t = (_Float16*)carve((size_t)768 * FFDIM * 2);
    _Float16* aggWt= (_Float16*)carve((size_t)768 * 768 * 2);
    _Float16* W0ct = (_Float16*)carve((size_t)512 * 768 * 2);
    _Float16* W0nt = (_Float16*)carve((size_t)512 * 768 * 2);
    _Float16* W1t  = (_Float16*)carve((size_t)128 * 512 * 2);
    float*    X32  = (float*)   carve((size_t)M1 * DMODEL * 4);
    _Float16* Xh   = (_Float16*)carve((size_t)M1 * DMODEL * 2);
    _Float16* Qh   = (_Float16*)carve((size_t)M1 * DMODEL * 2);
    _Float16* Kh   = (_Float16*)carve((size_t)M1 * DMODEL * 2);
    _Float16* Vh   = (_Float16*)carve((size_t)M1 * DMODEL * 2);
    _Float16* Oh   = (_Float16*)carve((size_t)M1 * DMODEL * 2);
    float*    TMP32= (float*)   carve((size_t)M1 * DMODEL * 4);
    _Float16* F1h  = (_Float16*)carve((size_t)FFCHUNK * FFDIM * 2);
    float*    X3pre= (float*)   carve((size_t)M2 * DMODEL * 4);
    float*    X3   = (float*)   carve((size_t)M2 * DMODEL * 4);
    _Float16* X3h  = (_Float16*)carve((size_t)M2 * DMODEL * 2);
    float*    A4   = (float*)   carve((size_t)M2 * 512 * 4);
    float*    B4   = (float*)   carve((size_t)M2 * 512 * 4);
    float*    cbuf = (float*)   carve((size_t)BS * 512 * 4);

    auto tp = [&](const float* W, _Float16* Wt_, int K, int N) {
        size_t n = (size_t)K * N;
        nb_transpose_f16<<<(unsigned)((n + 255) / 256), 256, 0, stream>>>(W, Wt_, K, N);
    };
    auto gemm = [&](const _Float16* A, const _Float16* Bt, const float* bias,
                    float* Cf, _Float16* Ch, int M, int N, int K, int act) {
        int blocks = ((M + BM - 1) / BM) * (N / BN);
        nb_gemm_wmma<<<blocks, 256, 0, stream>>>(A, Bt, bias, Cf, Ch, M, N, K, act);
    };

    // --- prep weights (f16, transposed to N x K) ---
    tp(Wq, Wqt, 768, 768);   tp(Wk, Wkt, 768, 768);
    tp(Wv, Wvt, 768, 768);   tp(Wo, Wot, 768, 768);
    tp(Wf1, Wf1t, 768, FFDIM);
    tp(Wf2, Wf2t, FFDIM, 768);
    tp(agg_W, aggWt, 768, 768);
    tp(W0,             W0ct, 768, 512);
    tp(W0 + 768 * 512, W0nt, 768, 512);
    tp(W1, W1t, 512, 128);

    // --- stage 1: token embedding + encoder over 32512 rows (S=64) ---
    nb_embed_ln<<<M1, 256, 0, stream>>>(ids, word_emb, pos_emb, ln_emb_g, ln_emb_b,
                                        X32, Xh, CLEN, 1);
    gemm(Xh, Wqt, bq, nullptr, Qh, M1, DMODEL, DMODEL, 0);
    gemm(Xh, Wkt, bk, nullptr, Kh, M1, DMODEL, DMODEL, 0);
    gemm(Xh, Wvt, bv, nullptr, Vh, M1, DMODEL, DMODEL, 0);
    nb_attention<<<BS * NC * NHEAD, CLEN, CLEN * DHEAD * 2 * sizeof(float), stream>>>(
        Qh, Kh, Vh, cell_masks, Oh, CLEN);
    gemm(Oh, Wot, bo, TMP32, nullptr, M1, DMODEL, DMODEL, 0);
    nb_add_ln<<<M1, 256, 0, stream>>>(X32, TMP32, ln1_g, ln1_b, X32, Xh, 0);
    for (int c = 0; c < M1 / FFCHUNK; ++c) {
        size_t off = (size_t)c * FFCHUNK;
        gemm(Xh + off * DMODEL, Wf1t, bf1, nullptr, F1h, FFCHUNK, FFDIM, DMODEL, 1);
        gemm(F1h, Wf2t, bf2, TMP32 + off * DMODEL, nullptr, FFCHUNK, DMODEL, FFDIM, 0);
    }
    nb_add_ln<<<M1, 256, 0, stream>>>(X32, TMP32, ln2_g, ln2_b, X32, Xh, 0);

    // --- stage 2: attention-pooled cells + start embedding ---
    gemm(Xh, aggWt, nullptr, TMP32, nullptr, M1, DMODEL, DMODEL, 2 /*tanh*/);
    nb_agg<<<BS * NC, CLEN, 0, stream>>>(TMP32, X32, agg_v, cell_masks, X3pre);
    nb_embed_ln<<<BS, 256, 0, stream>>>(start, word_emb, pos_emb, ln_emb_g, ln_emb_b,
                                        X3pre, nullptr, 0, S2);
    nb_add_ln<<<M2, 256, 0, stream>>>(X3pre, pos_emb, ln_emb_g, ln_emb_b, X3, X3h, S2);

    // --- stage 3: encoder over 512 rows (S=128) ---
    gemm(X3h, Wqt, bq, nullptr, Qh, M2, DMODEL, DMODEL, 0);
    gemm(X3h, Wkt, bk, nullptr, Kh, M2, DMODEL, DMODEL, 0);
    gemm(X3h, Wvt, bv, nullptr, Vh, M2, DMODEL, DMODEL, 0);
    nb_attention<<<BS * NHEAD, S2, S2 * DHEAD * 2 * sizeof(float), stream>>>(
        Qh, Kh, Vh, nb_masks, Oh, S2);
    gemm(Oh, Wot, bo, TMP32, nullptr, M2, DMODEL, DMODEL, 0);
    nb_add_ln<<<M2, 256, 0, stream>>>(X3, TMP32, ln1_g, ln1_b, X3, X3h, 0);
    gemm(X3h, Wf1t, bf1, nullptr, F1h, M2, FFDIM, DMODEL, 1);
    gemm(F1h, Wf2t, bf2, TMP32, nullptr, M2, DMODEL, FFDIM, 0);
    nb_add_ln<<<M2, 256, 0, stream>>>(X3, TMP32, ln2_g, ln2_b, X3, X3h, 0);

    // --- stage 4: pairwise head ---
    gemm(X3h, W0ct, nullptr, A4, nullptr, M2, 512, DMODEL, 0);
    gemm(X3h, W0nt, nullptr, B4, nullptr, M2, 512, DMODEL, 0);
    nb_cb<<<(BS * 512 + 255) / 256, 256, 0, stream>>>(fts, W0, b0, cbuf);
    nb_pairwise<<<dim3(8, NC, BS), 256, 0, stream>>>(A4, B4, cbuf, W1t, b1, Wt, bt,
                                                     label_mask, (float*)d_out);
}
